// GraphTransformerLayer_65687229825554
// MI455X (gfx1250) — compile-verified, matrix operand order checked by reading, not run on hardware
//
#include <hip/hip_runtime.h>

// Graph-transformer layer for MI455X (gfx1250, wave32, WMMA).
// N=4096, D=256, H=8, d_h=32, E=131072.
//
// bf16 WMMA (v_wmma_f32_16x16x32_bf16) for all GEMMs, fp32 accumulate.
// 2MB allowed-bitmask (L2-resident) for the graph mask.
// Flash-style masked softmax computed as S^T = K_A x Q_B so each lane owns a
// query row; exp/mask/repack into the PV A-operand is purely per-lane.
// Attention waves process 32 query rows (two Q tiles) so each K/V tile load
// feeds 2x the WMMAs. V is stored transposed (Vt[dv][m']) so PV B-operand
// loads are contiguous b128s, like the K path. Residual + LayerNorm in fp32.

#define NN 4096
#define DM 256
#define NH 8
#define DH 32
#define MASK_WORDS_PER_ROW (NN / 32) // 128

typedef __attribute__((ext_vector_type(16))) __bf16 v16bf;
typedef __attribute__((ext_vector_type(8)))  float  v8f;

__device__ __forceinline__ v8f wmma_bf16(v16bf a, v16bf b, v8f c) {
  // (neg_a, A, neg_b, B, c_mod, C, reuse_a, reuse_b)
  return __builtin_amdgcn_wmma_f32_16x16x32_bf16(false, a, false, b,
                                                 (short)0, c, false, false);
}

// Per-lane K-offset pattern for 16-bit A/B WMMA operands (gfx1250 wave32):
// VGPR j holds elements (k, k+1) with k = 2*(j&3) + 16*(j>>2) + 8*half,
// where half = lane/16 and the lane's row/col index is lane%16.
__device__ __forceinline__ int kpat(int j, int half) {
  return 2 * (j & 3) + ((j >> 2) << 4) + 8 * half;
}

// Row-major fp32 row -> bf16 operand (A row, or B column when B[k][n]=W[n][k]).
// Per lane this is two contiguous 32B spans; compiler merges to b128 loads.
__device__ __forceinline__ v16bf load_row_f32(const float* rowp, int k0, int half) {
  v16bf a;
#pragma unroll
  for (int j = 0; j < 8; ++j) {
    int k = k0 + kpat(j, half);
    float2 f = *(const float2*)(rowp + k);
    a[2 * j]     = (__bf16)f.x;
    a[2 * j + 1] = (__bf16)f.y;
  }
  return a;
}

// Same but source already bf16: two contiguous 16B spans per lane.
__device__ __forceinline__ v16bf load_row_bf16(const __bf16* rowp, int k0, int half) {
  v16bf a;
#pragma unroll
  for (int j = 0; j < 8; ++j) {
    int k = k0 + kpat(j, half);
    a[2 * j]     = rowp[k];
    a[2 * j + 1] = rowp[k + 1];
  }
  return a;
}

// ---------------- mask construction ----------------
// Bit (row m, col m') == 1  -> attention allowed (diagonal or edge dst->src).
__global__ __launch_bounds__(256) void k_mask_init(unsigned* mw) {
  int i = blockIdx.x * 256 + threadIdx.x;           // word index, N*N/32 total
  int row = i >> 7;                                 // / MASK_WORDS_PER_ROW
  int cw  = i & (MASK_WORDS_PER_ROW - 1);
  mw[i] = (cw == (row >> 5)) ? (1u << (row & 31)) : 0u;   // diagonal bit
}

__global__ __launch_bounds__(256) void k_mask_edges(const int* ei, unsigned* mw, int E) {
  int e = blockIdx.x * 256 + threadIdx.x;
  if (e < E) {
    int src = ei[e];        // edge_index[0][e]
    int dst = ei[E + e];    // edge_index[1][e]
    atomicOr(&mw[dst * MASK_WORDS_PER_ROW + (src >> 5)], 1u << (src & 31));
  }
}

// ---------------- QKV projection: P = x @ W^T (fp32 in, bf16 out) ----------
// Q,K stored row-major [m][d]; V stored transposed Vt[d][m] so the attention
// PV B-operand reads are contiguous.
__global__ __launch_bounds__(32) void k_qkv(const float* __restrict__ x,
                                            const float* __restrict__ Wq,
                                            const float* __restrict__ Wk,
                                            const float* __restrict__ Wv,
                                            unsigned short* Qb_,
                                            unsigned short* Kb_,
                                            unsigned short* Vt_) {
  const int lane = threadIdx.x, ln = lane & 15, half = lane >> 4;
  const int it = blockIdx.x;            // 16-row tile of x
  const int nt = blockIdx.y;            // 16-col tile of output
  const int which = blockIdx.z;         // 0:Q 1:K 2:V
  const float* W = (which == 0) ? Wq : (which == 1) ? Wk : Wv;

  const float* arow = x + (it * 16 + ln) * DM;      // A: row m of x
  const float* brow = W + (nt * 16 + ln) * DM;      // B[k][n] = W[n][k]
  v8f c = {};
#pragma unroll
  for (int k0 = 0; k0 < DM; k0 += 32)
    c = wmma_bf16(load_row_f32(arow, k0, half), load_row_f32(brow, k0, half), c);

  if (which < 2) {
    __bf16* P = (__bf16*)((which == 0) ? Qb_ : Kb_);
#pragma unroll
    for (int r = 0; r < 8; ++r) {
      int row = it * 16 + r + 8 * half; // D layout: VGPR r -> M = r + 8*(lane/16)
      P[row * DM + nt * 16 + ln] = (__bf16)c[r];
    }
  } else {
    __bf16* Vt = (__bf16*)Vt_;          // Vt[d][m], row stride NN
#pragma unroll
    for (int r = 0; r < 8; ++r) {
      int row = it * 16 + r + 8 * half;
      Vt[(nt * 16 + ln) * NN + row] = (__bf16)c[r];
    }
  }
}

// ------- masked flash attention: per head, per 32-row Q block, one wave -----
__global__ __launch_bounds__(32) void k_attn(const unsigned* __restrict__ mw,
                                             const unsigned short* Qb_,
                                             const unsigned short* Kb_,
                                             const unsigned short* Vt_,
                                             unsigned short* Ob_) {
  const __bf16* Qb = (const __bf16*)Qb_;
  const __bf16* Kb = (const __bf16*)Kb_;
  const __bf16* Vt = (const __bf16*)Vt_;
  __bf16* Ob = (__bf16*)Ob_;

  __shared__ float ssum[32];
  const int lane = threadIdx.x, ln = lane & 15, half = lane >> 4;
  const int it   = blockIdx.x;                  // 32-row query block
  const int head = blockIdx.y;
  const int hoff = head * DH;
  const int ma   = it * 32 + ln;                // lane's query row, tile a
  const int mb   = ma + 16;                     // lane's query row, tile b
  const float sc = 0.17677669529663689f;        // 1/sqrt(32)

  // Q as B operand of S^T = K_A x Q_B: lane holds column m, contiguous d_h.
  const v16bf qba = load_row_bf16(Qb + ma * DM + hoff, 0, half);
  const v16bf qbb = load_row_bf16(Qb + mb * DM + hoff, 0, half);
  const unsigned* mra = mw + ma * MASK_WORDS_PER_ROW;
  const unsigned* mrb = mw + mb * MASK_WORDS_PER_ROW;

  v8f o0a = {}, o1a = {}, o0b = {}, o1b = {};   // O: {tile a,b} x {dv lo,hi}
  float rsa = 0.f, rsb = 0.f;                   // per-lane half row-sums

  for (int j0 = 0; j0 < NN; j0 += 32) {
    if (j0 + 32 < NN) {
      __builtin_prefetch(Kb + (j0 + 32 + ln) * DM + hoff, 0, 0);
      __builtin_prefetch(Vt + (hoff + ln) * NN + j0 + 32, 0, 0);
    }
    // one mask word covers exactly m' = j0..j0+31 for each query row
    const unsigned wa = mra[j0 >> 5];
    const unsigned wb = mrb[j0 >> 5];

    // K tiles shared by both Q tiles. D lane l: col = query row, reg r:
    // m' = j0 (+16) + r + 8*half.
    const v16bf ka0 = load_row_bf16(Kb + (j0 + ln) * DM + hoff, 0, half);
    const v16bf ka1 = load_row_bf16(Kb + (j0 + 16 + ln) * DM + hoff, 0, half);
    v8f s0a = {}, s1a = {}, s0b = {}, s1b = {};
    s0a = wmma_bf16(ka0, qba, s0a);
    s1a = wmma_bf16(ka1, qba, s1a);
    s0b = wmma_bf16(ka0, qbb, s0b);
    s1b = wmma_bf16(ka1, qbb, s1b);

    // exp + mask; per-lane repack into PV A-operands: lane row = m, and the
    // lane's D-layout m' set {8h..8h+7} (+16) is exactly its A k-set.
    v16bf paa, pab;
#pragma unroll
    for (int r = 0; r < 8; ++r) {
      int mp = r + 8 * half;
      float pa0 = ((wa >> mp) & 1u)        ? __expf(s0a[r] * sc) : 0.f;
      float pa1 = ((wa >> (16 + mp)) & 1u) ? __expf(s1a[r] * sc) : 0.f;
      float pb0 = ((wb >> mp) & 1u)        ? __expf(s0b[r] * sc) : 0.f;
      float pb1 = ((wb >> (16 + mp)) & 1u) ? __expf(s1b[r] * sc) : 0.f;
      rsa += pa0 + pa1;
      rsb += pb0 + pb1;
      paa[r] = (__bf16)pa0;  paa[8 + r] = (__bf16)pa1;
      pab[r] = (__bf16)pb0;  pab[8 + r] = (__bf16)pb1;
    }

    // O += P_A x V_B ; V from Vt[d][m']: lane column dv -> contiguous m'.
    const v16bf vb0 = load_row_bf16(Vt + (hoff + ln) * NN + j0, 0, half);
    const v16bf vb1 = load_row_bf16(Vt + (hoff + 16 + ln) * NN + j0, 0, half);
    o0a = wmma_bf16(paa, vb0, o0a);
    o1a = wmma_bf16(paa, vb1, o1a);
    o0b = wmma_bf16(pab, vb0, o0b);
    o1b = wmma_bf16(pab, vb1, o1b);
  }

  // merge the two half-row sums (lanes l and l^16 hold the same query row)
  rsa += __shfl_xor(rsa, 16);
  rsb += __shfl_xor(rsb, 16);
  if (half == 0) { ssum[ln] = rsa; ssum[16 + ln] = rsb; }
  __syncthreads();

#pragma unroll
  for (int r = 0; r < 8; ++r) {
    int mr = r + 8 * half;                      // O rows: m = r + 8*(lane/16)
    float inva = 1.f / ssum[mr];
    float invb = 1.f / ssum[16 + mr];
    int rowa = it * 32 + mr;
    int rowb = rowa + 16;
    Ob[rowa * DM + hoff + ln]      = (__bf16)(o0a[r] * inva);
    Ob[rowa * DM + hoff + 16 + ln] = (__bf16)(o1a[r] * inva);
    Ob[rowb * DM + hoff + ln]      = (__bf16)(o0b[r] * invb);
    Ob[rowb * DM + hoff + 16 + ln] = (__bf16)(o1b[r] * invb);
  }
}

// ---------------- output projection + bias + residual (fp32 out) ----------
__global__ __launch_bounds__(32) void k_oproj(const unsigned short* Ob_,
                                              const float* __restrict__ Wo,
                                              const float* __restrict__ bo,
                                              const float* __restrict__ x,
                                              float* __restrict__ hbuf) {
  const __bf16* Ob = (const __bf16*)Ob_;
  const int lane = threadIdx.x, ln = lane & 15, half = lane >> 4;
  const int it = blockIdx.x, nt = blockIdx.y;

  const __bf16* arow = Ob + (it * 16 + ln) * DM;
  const float*  brow = Wo + (nt * 16 + ln) * DM;  // B[k][n] = Wo[n][k]
  v8f c = {};
#pragma unroll
  for (int k0 = 0; k0 < DM; k0 += 32)
    c = wmma_bf16(load_row_bf16(arow, k0, half), load_row_f32(brow, k0, half), c);

#pragma unroll
  for (int r = 0; r < 8; ++r) {
    int row = it * 16 + r + 8 * half;
    int col = nt * 16 + ln;
    hbuf[row * DM + col] = c[r] + bo[col] + x[row * DM + col];
  }
}

// ---------------- LayerNorm over D=256 per row ----------------
__global__ __launch_bounds__(256) void k_ln(const float* __restrict__ hbuf,
                                            const float* __restrict__ gamma,
                                            const float* __restrict__ beta,
                                            float* __restrict__ out) {
  __shared__ float red[2][8];
  const int t = threadIdx.x, row = blockIdx.x;
  const float v = hbuf[row * DM + t];
  float s = v, q = v * v;
#pragma unroll
  for (int o = 16; o > 0; o >>= 1) { s += __shfl_xor(s, o); q += __shfl_xor(q, o); }
  if ((t & 31) == 0) { red[0][t >> 5] = s; red[1][t >> 5] = q; }
  __syncthreads();
  float S = 0.f, Q2 = 0.f;
#pragma unroll
  for (int i = 0; i < 8; ++i) { S += red[0][i]; Q2 += red[1][i]; }
  const float mu  = S * (1.0f / DM);
  const float var = Q2 * (1.0f / DM) - mu * mu;
  const float inv = rsqrtf(var + 1e-5f);
  out[row * DM + t] = (v - mu) * inv * gamma[t] + beta[t];
}

extern "C" void kernel_launch(void* const* d_in, const int* in_sizes, int n_in,
                              void* d_out, int out_size, void* d_ws, size_t ws_size,
                              hipStream_t stream) {
  const float* x     = (const float*)d_in[0];
  const int*   ei    = (const int*)d_in[1];
  const float* Wq    = (const float*)d_in[2];
  const float* Wk    = (const float*)d_in[3];
  const float* Wv    = (const float*)d_in[4];
  const float* Wo    = (const float*)d_in[5];
  const float* bo    = (const float*)d_in[6];
  const float* gamma = (const float*)d_in[7];
  const float* beta  = (const float*)d_in[8];
  float* out = (float*)d_out;
  const int E = in_sizes[1] / 2;

  // workspace layout (bytes): mask 2MB | Qb 2MB | Kb 2MB | Vt 2MB | Ob 2MB | h 4MB
  char* ws = (char*)d_ws;
  unsigned*       maskw = (unsigned*)ws;
  unsigned short* Qb = (unsigned short*)(ws + (size_t)2 * 1024 * 1024);
  unsigned short* Kb = (unsigned short*)(ws + (size_t)4 * 1024 * 1024);
  unsigned short* Vt = (unsigned short*)(ws + (size_t)6 * 1024 * 1024);
  unsigned short* Ob = (unsigned short*)(ws + (size_t)8 * 1024 * 1024);
  float*          hb = (float*)(ws + (size_t)10 * 1024 * 1024);

  k_mask_init<<<(NN * (NN / 32)) / 256, 256, 0, stream>>>(maskw);
  k_mask_edges<<<(E + 255) / 256, 256, 0, stream>>>(ei, maskw, E);

  dim3 gq(NN / 16, DM / 16, 3);
  k_qkv<<<gq, 32, 0, stream>>>(x, Wq, Wk, Wv, Qb, Kb, Vt);

  dim3 ga(NN / 32, NH);
  k_attn<<<ga, 32, 0, stream>>>(maskw, Qb, Kb, Vt, Ob);

  dim3 go(NN / 16, DM / 16);
  k_oproj<<<go, 32, 0, stream>>>(Ob, Wo, bo, x, hb);

  k_ln<<<NN, 256, 0, stream>>>(hb, gamma, beta, out);
}
